// ModulatedConv2d_2dStyleTensor_88210038325479
// MI455X (gfx1250) — matrix operand order repaired and checked by reading, not compiled, not run on hardware
//
#include <hip/hip_runtime.h>

#define BB   8
#define CIN  256
#define COUT 256
#define SD   256
#define HH   128
#define WW   128
#define HWP  (HH*WW)   // 16384

typedef __attribute__((ext_vector_type(16))) __bf16 v16bf;
typedef __attribute__((ext_vector_type(8)))  float  v8f;
typedef int v4i_n __attribute__((vector_size(16)));   // matches async builtin param

#if defined(__gfx1250__) && __has_builtin(__builtin_amdgcn_global_load_async_to_lds_b128)
#define USE_ASYNC 1
#else
#define USE_ASYNC 0
#endif

__device__ __forceinline__ void wait_async0() {
#if __has_builtin(__builtin_amdgcn_s_wait_asynccnt)
    __builtin_amdgcn_s_wait_asynccnt(0);
#else
    asm volatile("s_wait_asynccnt 0x0" ::: "memory");
#endif
}

#if USE_ASYNC
__device__ __forceinline__ void async_b128(const unsigned short* gsrc,
                                           unsigned short* ldst) {
    auto g = (__attribute__((address_space(1))) v4i_n*)
                 (unsigned long long)(const void*)gsrc;
    auto l = (__attribute__((address_space(3))) v4i_n*)
                 (unsigned int)(unsigned long long)(void*)ldst;
    __builtin_amdgcn_global_load_async_to_lds_b128(g, l, 0, 0);
}
#endif

// round-to-nearest-even f32 -> bf16 bits
__device__ __forceinline__ unsigned short f2bf(float f) {
    unsigned int u = __float_as_uint(f);
    u += 0x7FFFu + ((u >> 16) & 1u);
    return (unsigned short)(u >> 16);
}

// XOR-swizzled LDS index for a 32-element bf16 row split in 8-element groups:
// row r, group g (0..3) -> element index. Spreads the 64B row stride of
// fragment reads across all 64 LDS banks.
__device__ __forceinline__ int swz(int r, int g) {
    return r * 32 + (((g ^ (r & 3) ^ ((r >> 2) & 3)) & 3) << 3);
}

// build a 16-element bf16 fragment from two 16-byte LDS chunks
__device__ __forceinline__ v16bf load_frag16(const unsigned short* b0,
                                             const unsigned short* b1) {
    v16bf r;
    uint4* p = reinterpret_cast<uint4*>(&r);
    p[0] = *reinterpret_cast<const uint4*>(b0);
    p[1] = *reinterpret_cast<const uint4*>(b1);
    return r;
}

// ---------------------------------------------------------------------------
// Kernel 1: style 1x1 conv GEMM (M=CIN, K=SD, N=64 pixels) via bf16 WMMA.
// Epilogue: s = acc*sc_scale + bias; xin = bf16(x*s); atomic s-mean partials.
// ---------------------------------------------------------------------------
__global__ __launch_bounds__(256) void style_mod_kernel(
    const float* __restrict__ x, const float* __restrict__ style,
    const float* __restrict__ sc_w, const float* __restrict__ sc_b,
    unsigned short* __restrict__ xin, float* __restrict__ smean)
{
    __shared__ unsigned short ldsA[256 * 32]; // [cin][k] swizzled
    __shared__ unsigned short ldsB[64 * 32];  // [n][k]   swizzled

    const int tid  = threadIdx.x;
    const int lane = tid & 31;
    const int wv   = tid >> 5;
    const int b    = blockIdx.x >> 8;
    const int p0   = (blockIdx.x & 255) * 64;
    const int m0   = wv * 32;

    const v8f vzero = {};
    v8f acc[2][4];
    #pragma unroll
    for (int i = 0; i < 2; ++i)
        #pragma unroll
        for (int j = 0; j < 4; ++j) acc[i][j] = vzero;

    const int nlo   = lane & 15;
    const int akoff = (lane < 16) ? 0 : 8;   // logical K group base for A
    const int bkoff = (lane < 16) ? 0 : 8;   // logical K group base for B

    for (int kc = 0; kc < SD / 32; ++kc) {
        // A tile: sc_weight (reused by all blocks -> keep temporal)
        {
            const float* src = sc_w + tid * SD + kc * 32;
            #pragma unroll
            for (int j = 0; j < 32; j += 4) {
                float4 v = *reinterpret_cast<const float4*>(src + j);
                int base = swz(tid, j >> 3) + (j & 7);
                ldsA[base + 0] = f2bf(v.x); ldsA[base + 1] = f2bf(v.y);
                ldsA[base + 2] = f2bf(v.z); ldsA[base + 3] = f2bf(v.w);
            }
        }
        // B tile: style (streamed once -> non-temporal)
        {
            const int n  = tid >> 2;
            const int kb = (tid & 3) * 8;
            union { unsigned short s[8]; uint4 v; } tmp;
            #pragma unroll
            for (int j = 0; j < 8; ++j) {
                int sd = kc * 32 + kb + j;
                tmp.s[j] = f2bf(__builtin_nontemporal_load(
                    &style[(b * SD + sd) * HWP + p0 + n]));
            }
            *reinterpret_cast<uint4*>(&ldsB[swz(n, kb >> 3)]) = tmp.v;
        }
        __syncthreads();

        v16bf afrag[2];
        #pragma unroll
        for (int mt = 0; mt < 2; ++mt) {
            int row = m0 + mt * 16 + nlo;
            afrag[mt] = load_frag16(&ldsA[swz(row, (akoff >> 3) + 0)],
                                    &ldsA[swz(row, (akoff >> 3) + 2)]);
        }
        #pragma unroll
        for (int nt = 0; nt < 4; ++nt) {
            int col = nt * 16 + nlo;
            v16bf bfrag = load_frag16(&ldsB[swz(col, (bkoff >> 3) + 0)],
                                      &ldsB[swz(col, (bkoff >> 3) + 2)]);
            #pragma unroll
            for (int mt = 0; mt < 2; ++mt)
                acc[mt][nt] = __builtin_amdgcn_wmma_f32_16x16x32_bf16(
                    false, afrag[mt], false, bfrag, (short)0, acc[mt][nt],
                    false, false);
        }
        __syncthreads();
    }

    // Epilogue: bias+scale, modulate x, write bf16 xin, reduce s for mean
    const float sc_scale = 0.0625f;          // 1/sqrt(SD)
    const int   moff     = (lane < 16) ? 0 : 8;
    #pragma unroll
    for (int mt = 0; mt < 2; ++mt) {
        #pragma unroll
        for (int r = 0; r < 8; ++r) {
            int c = m0 + mt * 16 + r + moff;
            float bias = sc_b[c];
            float rowsum = 0.f;
            #pragma unroll
            for (int nt = 0; nt < 4; ++nt) {
                float s = acc[mt][nt][r] * sc_scale + bias;
                rowsum += s;
                int p  = p0 + nt * 16 + nlo;
                int gi = (b * CIN + c) * HWP + p;
                float xv = __builtin_nontemporal_load(&x[gi]); // x read once
                xin[gi] = f2bf(xv * s);  // xin re-read many times -> L2 (RT)
            }
            #pragma unroll
            for (int m = 8; m >= 1; m >>= 1)
                rowsum += __shfl_xor(rowsum, m, 16);
            if (nlo == 0) atomicAdd(&smean[b * CIN + c], rowsum);
        }
    }
}

// ---------------------------------------------------------------------------
// Kernel 2: demod = rsqrt(sum((scale*w*smean)^2)+eps); write bf16 weights in
// tap-major layout [b][kk][cout][cin] with the final output scale folded in.
// ---------------------------------------------------------------------------
__global__ __launch_bounds__(256) void demod_kernel(
    const float* __restrict__ mw, const float* __restrict__ smean,
    unsigned short* __restrict__ wgt)
{
    __shared__ float red[256];
    const int tid = threadIdx.x;
    const int b   = blockIdx.x >> 8;
    const int co  = blockIdx.x & 255;
    const float scale = 1.0f / 48.0f;        // 1/sqrt(CIN*3*3)
    const float invHW = 1.0f / 16384.0f;

    float partial = 0.f;
    for (int idx = tid; idx < CIN * 9; idx += 256) {
        int cin = idx / 9;
        int kk  = idx - cin * 9;
        float w  = mw[(co * CIN + cin) * 9 + kk];
        float sm = smean[b * CIN + cin] * invHW;
        float v  = scale * w * sm;
        partial += v * v;
    }
    red[tid] = partial;
    __syncthreads();
    for (int s2 = 128; s2 > 0; s2 >>= 1) {
        if (tid < s2) red[tid] += red[tid + s2];
        __syncthreads();
    }
    float demod = rsqrtf(red[0] + 1e-8f);
    const float outscale = (1.0f / 48.0f) * (1.0f / 128.0f); // scale*style_scale
    float wsc = scale * demod * outscale;

    for (int idx = tid; idx < CIN * 9; idx += 256) {
        int cin = idx / 9;
        int kk  = idx - cin * 9;
        float w = mw[(co * CIN + cin) * 9 + kk];
        wgt[((b * 9 + kk) * COUT + co) * CIN + cin] = f2bf(w * wsc);
    }
}

// ---------------------------------------------------------------------------
// Kernel 3: per-sample 3x3 conv as implicit GEMM, double-buffered LDS.
// M=COUT, K=CIN*9, N=HW. BM=128, BN=128, BK=32; 72 K-iterations.
// Wave tile 32x64 -> 8 WMMA per K-step. A tile via async global->LDS DMA
// (ASYNCcnt); B tile via VGPRs (zero-fill halo). One barrier per K-step.
// ---------------------------------------------------------------------------
__global__ __launch_bounds__(256) void conv_wmma_kernel(
    const unsigned short* __restrict__ xin,
    const unsigned short* __restrict__ wgt,
    float* __restrict__ out)
{
    __shared__ unsigned short ldsA[2][128 * 32]; // [m][k] ping-pong, swizzled
    __shared__ unsigned short ldsB[2][128 * 32]; // [n][k] ping-pong, swizzled

    const int tid  = threadIdx.x;
    const int lane = tid & 31;
    const int wv   = tid >> 5;
    const int wm   = wv & 3;                 // 4 m-stripes of 32
    const int wn   = wv >> 2;                // 2 n-stripes of 64

    const int nblk = blockIdx.x & 127;       // HW/128
    const int mblk = (blockIdx.x >> 7) & 1;  // COUT/128
    const int b    = blockIdx.x >> 8;

    const int p0  = nblk * 128;              // pixel tile = one image row
    const int y0  = nblk;
    const int m0g = mblk * 128;

    // producer coordinates (fixed per thread)
    const int arow = tid >> 1;
    const int acb  = (tid & 1) * 16;         // A: 16 contiguous k per thread
    const int bn   = tid >> 1;               // B: 128 rows, 2 threads per row
    const int bkb  = (tid & 1) * 16;         // B: 16 k (strided) per thread

    const v8f vzero = {};
    v8f acc[2][4];
    #pragma unroll
    for (int i = 0; i < 2; ++i)
        #pragma unroll
        for (int j = 0; j < 4; ++j) acc[i][j] = vzero;

    const int nlo   = lane & 15;
    const int akoff = (lane < 16) ? 0 : 8;
    const int bkoff = (lane < 16) ? 0 : 8;

    auto load_tiles = [&](int it, int buf) {
        const int kk = it >> 3;              // tap index 0..8
        const int cc = it & 7;               // cin chunk 0..7
        const int ky = kk / 3 - 1;
        const int kx = (kk - (kk / 3) * 3) - 1;
        const int py = y0 + ky;
        const bool yok = (py >= 0) && (py < HH);

        // ---- A tile: 128x32 bf16, contiguous; swizzled DMA dests ----
        const unsigned short* asrc =
            wgt + ((size_t)(b * 9 + kk) * COUT + m0g + arow) * CIN + cc * 32 + acb;
        unsigned short* ad0 = &ldsA[buf][swz(arow, (acb >> 3) + 0)];
        unsigned short* ad1 = &ldsA[buf][swz(arow, (acb >> 3) + 1)];
#if USE_ASYNC
        async_b128(asrc,     ad0);
        async_b128(asrc + 8, ad1);
#else
        *reinterpret_cast<uint4*>(ad0) = *reinterpret_cast<const uint4*>(asrc);
        *reinterpret_cast<uint4*>(ad1) = *reinterpret_cast<const uint4*>(asrc + 8);
#endif
        // ---- B tile: xin with tap shift + zero-fill halo ----
        const int px = bn + kx;              // x0 == 0
        const bool ok = yok && (px >= 0) && (px < WW);
        union { unsigned short s[8]; uint4 v; } t0, t1;
        if (ok) {
            const unsigned short* src =
                xin + (size_t)(b * CIN + cc * 32 + bkb) * HWP + py * WW + px;
            #pragma unroll
            for (int j = 0; j < 8; ++j) {
                t0.s[j] = src[(size_t)j * HWP];
                t1.s[j] = src[(size_t)(j + 8) * HWP];
            }
        } else {
            #pragma unroll
            for (int j = 0; j < 8; ++j) { t0.s[j] = 0; t1.s[j] = 0; }
        }
        *reinterpret_cast<uint4*>(&ldsB[buf][swz(bn, (bkb >> 3) + 0)]) = t0.v;
        *reinterpret_cast<uint4*>(&ldsB[buf][swz(bn, (bkb >> 3) + 1)]) = t1.v;
    };

    // prologue: fill buffer 0
    load_tiles(0, 0);
#if USE_ASYNC
    wait_async0();
#endif
    __syncthreads();

    #pragma unroll 2
    for (int it = 0; it < 72; ++it) {
        const int cur = it & 1;

        // consumer fragment reads first (DS queue ahead of next producers)
        v16bf afrag[2], bfrag[4];
        #pragma unroll
        for (int mt = 0; mt < 2; ++mt) {
            int row = wm * 32 + mt * 16 + nlo;
            afrag[mt] = load_frag16(&ldsA[cur][swz(row, (akoff >> 3) + 0)],
                                    &ldsA[cur][swz(row, (akoff >> 3) + 2)]);
        }
        #pragma unroll
        for (int nt = 0; nt < 4; ++nt) {
            int col = wn * 64 + nt * 16 + nlo;
            bfrag[nt] = load_frag16(&ldsB[cur][swz(col, (bkoff >> 3) + 0)],
                                    &ldsB[cur][swz(col, (bkoff >> 3) + 2)]);
        }

        // prefetch next tile into the other buffer while WMMAs run
        if (it + 1 < 72) load_tiles(it + 1, cur ^ 1);

        #pragma unroll
        for (int mt = 0; mt < 2; ++mt)
            #pragma unroll
            for (int nt = 0; nt < 4; ++nt)
                acc[mt][nt] = __builtin_amdgcn_wmma_f32_16x16x32_bf16(
                    false, afrag[mt], false, bfrag[nt], (short)0,
                    acc[mt][nt], false, false);

        if (it + 1 < 72) {
#if USE_ASYNC
            wait_async0();
#endif
            __syncthreads();
        }
    }

    // store f32 output (final scale folded into weights); written once -> NT
    const int moff = (lane < 16) ? 0 : 8;
    #pragma unroll
    for (int mt = 0; mt < 2; ++mt) {
        #pragma unroll
        for (int nt = 0; nt < 4; ++nt) {
            int colg = p0 + wn * 64 + nt * 16 + nlo;
            #pragma unroll
            for (int r = 0; r < 8; ++r) {
                int rowg = m0g + wm * 32 + mt * 16 + r + moff;
                __builtin_nontemporal_store(
                    acc[mt][nt][r], &out[(size_t)(b * COUT + rowg) * HWP + colg]);
            }
        }
    }
}

// ---------------------------------------------------------------------------
extern "C" void kernel_launch(void* const* d_in, const int* in_sizes, int n_in,
                              void* d_out, int out_size, void* d_ws, size_t ws_size,
                              hipStream_t stream)
{
    const float* x     = (const float*)d_in[0];
    const float* style = (const float*)d_in[1];
    const float* mw    = (const float*)d_in[2];
    const float* scw   = (const float*)d_in[3];
    const float* scb   = (const float*)d_in[4];
    float* out = (float*)d_out;

    unsigned char* ws = (unsigned char*)d_ws;
    unsigned short* xin = (unsigned short*)ws;                       // 64 MB bf16
    size_t off = (size_t)BB * CIN * HWP * sizeof(unsigned short);
    unsigned short* wgt = (unsigned short*)(ws + off);               // 9 MB bf16
    off += (size_t)BB * 9 * COUT * CIN * sizeof(unsigned short);
    float* smean = (float*)(ws + off);                               // 8 KB

    (void)hipMemsetAsync(smean, 0, (size_t)BB * CIN * sizeof(float), stream);
    style_mod_kernel<<<BB * (HWP / 64), 256, 0, stream>>>(x, style, scw, scb,
                                                          xin, smean);
    demod_kernel<<<BB * COUT, 256, 0, stream>>>(mw, smean, wgt);
    conv_wmma_kernel<<<BB * 2 * (HWP / 128), 256, 0, stream>>>(xin, wgt, out);
}
